// TransformerDecoder_53300544143915
// MI455X (gfx1250) — matrix operand order repaired and checked by reading, not compiled, not run on hardware
//
#include <hip/hip_runtime.h>
#include <hip/hip_bf16.h>
#include <math.h>

// ---------------------------------------------------------------------------
// TransformerDecoder for MI455X (gfx1250).
// All GEMMs: bf16 x bf16 -> f32 via v_wmma_f32_16x16x32_bf16, double-buffered
// LDS tiles filled by GLOBAL_LOAD_ASYNC_TO_LDS_B128 (ASYNCcnt pipeline) when
// the builtin is available. K/V cross tensors kept bf16 (67MB -> L2 resident).
// Attention softmax in f32 VALU. Workspace requirement: ~305 MB.
// ---------------------------------------------------------------------------

typedef __bf16 bf16_t;
typedef __attribute__((ext_vector_type(16))) __bf16 v16bf;
typedef __attribute__((ext_vector_type(8)))  float  v8f;
typedef __attribute__((ext_vector_type(4)))  int    v4i;

#define AS1 __attribute__((address_space(1)))
#define AS3 __attribute__((address_space(3)))

#if defined(__AMDGCN__) && __has_builtin(__builtin_amdgcn_global_load_async_to_lds_b128)
#define GEMM_ASYNC 1
#endif

static __device__ __forceinline__ void wait_async0() {
#if defined(__AMDGCN__)
#if __has_builtin(__builtin_amdgcn_s_wait_asynccnt)
  __builtin_amdgcn_s_wait_asynccnt(0);
#else
  asm volatile("s_wait_asynccnt 0x0" ::: "memory");
#endif
#endif
}

constexpr int Bc   = 128;
constexpr int NQc  = 18;
constexpr int HWc  = 1024;
constexpr int Dc   = 256;
constexpr int Hc   = 8;
constexpr int DHc  = 32;
constexpr int Lc   = 6;
constexpr int HIDc = 1024;
constexpr int MQ   = Bc * NQc;   // 2304 query rows
constexpr int MK   = Bc * HWc;   // 131072 memory rows
constexpr float SCALEc = 0.17677669529663687f;  // 32^-0.5
constexpr float NEGINF = -1e30f;

// bf16 pair (packed in u32) -> two floats, cheap bit tricks
static __device__ __forceinline__ void unpack2(unsigned u, float* o) {
  o[0] = __uint_as_float(u << 16);
  o[1] = __uint_as_float(u & 0xffff0000u);
}

// ---------------------------------------------------------------------------
// vectorized converts (n multiple of 4)
// ---------------------------------------------------------------------------
union PK4 { bf16_t h[4]; uint2 u; };

__global__ void k_cvt4(bf16_t* __restrict__ dst, const float* __restrict__ src, int n4) {
  int i = blockIdx.x * blockDim.x + threadIdx.x;
  if (i >= n4) return;
  float4 f = ((const float4*)src)[i];
  PK4 o;
  o.h[0] = (bf16_t)f.x; o.h[1] = (bf16_t)f.y; o.h[2] = (bf16_t)f.z; o.h[3] = (bf16_t)f.w;
  ((uint2*)dst)[i] = o.u;
}

__global__ void k_add_cvt4(bf16_t* __restrict__ dst, const float* __restrict__ a,
                           const float* __restrict__ b, int n4) {
  int i = blockIdx.x * blockDim.x + threadIdx.x;
  if (i >= n4) return;
  float4 fa = ((const float4*)a)[i];
  float4 fb = ((const float4*)b)[i];
  PK4 o;
  o.h[0] = (bf16_t)(fa.x + fb.x); o.h[1] = (bf16_t)(fa.y + fb.y);
  o.h[2] = (bf16_t)(fa.z + fb.z); o.h[3] = (bf16_t)(fa.w + fb.w);
  ((uint2*)dst)[i] = o.u;
}

// ---------------------------------------------------------------------------
// transpose-convert: src f32 [z][R][C] -> dst bf16 [z][C][R]  (one-time)
// ---------------------------------------------------------------------------
__global__ void k_transpose(bf16_t* __restrict__ dst, const float* __restrict__ src,
                            int R, int C) {
  __shared__ float t[32][33];
  size_t zo = (size_t)blockIdx.z * R * C;
  src += zo; dst += zo;
  int c0 = blockIdx.x * 32, r0 = blockIdx.y * 32;
  int tx = threadIdx.x, ty = threadIdx.y;  // 32 x 8
  for (int i = ty; i < 32; i += 8) {
    int r = r0 + i, c = c0 + tx;
    t[i][tx] = (r < R && c < C) ? src[(size_t)r * C + c] : 0.f;
  }
  __syncthreads();
  for (int i = ty; i < 32; i += 8) {
    int c = c0 + i, r = r0 + tx;
    if (c < C && r < R) dst[(size_t)c * R + r] = (bf16_t)t[tx][i];
  }
}

// ---------------------------------------------------------------------------
// WMMA GEMM: C[M,N] = epi( A[M,K](bf16) * Bt[N,K]^T(bf16) + bias[N] )
// 64x64 tile / 128 threads (4 waves, each 32x32 = 2x2 wmma frags).
// Double-buffered LDS; async global->LDS DMA (ASYNCcnt) when available.
// Fragment per-lane layout (CDNA5 ISA 7.12.2):
//   A 16x32 : lane<16 -> M=lane, K in {0..7}U{16..23}; lane>=16 -> K {8..15}U{24..31}
//   B 32x16 : lane<16 -> N=lane, K in {0..7}U{16..23}; lane>=16 -> K {8..15}U{24..31} (same as A)
// ---------------------------------------------------------------------------
union ABu { unsigned int u[8]; v16bf v; };

template<bool GELU>
__global__ __launch_bounds__(128)
void k_gemm(const bf16_t* __restrict__ A, const bf16_t* __restrict__ Bt,
            const float* __restrict__ bias,
            float* __restrict__ Cf, bf16_t* __restrict__ Cb,
            int M, int N, int K) {
  __shared__ unsigned int lsA[2][64][17];  // packed bf16 pairs along K, +1 pad
  __shared__ unsigned int lsB[2][64][17];

  const int tid  = threadIdx.x;
  const int wave = tid >> 5, lane = tid & 31;
  const int wm = (wave >> 1) * 32, wn = (wave & 1) * 32;
  const int r0 = lane & 15, hi = lane >> 4;
  const int m0 = blockIdx.y * 64, n0 = blockIdx.x * 64;
  const int row = tid >> 1, half = tid & 1;

  const unsigned int* Ag = (const unsigned int*)(A + (size_t)(m0 + row) * K) + half * 8;
  const unsigned int* Bg = (const unsigned int*)(Bt + (size_t)(n0 + row) * K) + half * 8;

  v8f acc[2][2] = {};

  auto issue = [&](int buf, int k0) {
    const unsigned int* ap = Ag + (k0 >> 1);
    const unsigned int* bp = Bg + (k0 >> 1);
    unsigned int* as = &lsA[buf][row][half * 8];
    unsigned int* bs = &lsB[buf][row][half * 8];
#ifdef GEMM_ASYNC
    // IOFFSET is added to both the global and the LDS address (ISA 10.7/4.4)
    __builtin_amdgcn_global_load_async_to_lds_b128((AS1 v4i*)(void*)(unsigned int*)ap,
                                                   (AS3 v4i*)as, 0, 0);
    __builtin_amdgcn_global_load_async_to_lds_b128((AS1 v4i*)(void*)(unsigned int*)ap,
                                                   (AS3 v4i*)as, 16, 0);
    __builtin_amdgcn_global_load_async_to_lds_b128((AS1 v4i*)(void*)(unsigned int*)bp,
                                                   (AS3 v4i*)bs, 0, 0);
    __builtin_amdgcn_global_load_async_to_lds_b128((AS1 v4i*)(void*)(unsigned int*)bp,
                                                   (AS3 v4i*)bs, 16, 0);
#else
    const uint4 a0 = *(const uint4*)ap;
    const uint4 a1 = *(const uint4*)(ap + 4);
    const uint4 b0 = *(const uint4*)bp;
    const uint4 b1 = *(const uint4*)(bp + 4);
    as[0]=a0.x; as[1]=a0.y; as[2]=a0.z; as[3]=a0.w;
    as[4]=a1.x; as[5]=a1.y; as[6]=a1.z; as[7]=a1.w;
    bs[0]=b0.x; bs[1]=b0.y; bs[2]=b0.z; bs[3]=b0.w;
    bs[4]=b1.x; bs[5]=b1.y; bs[6]=b1.z; bs[7]=b1.w;
#endif
  };

  issue(0, 0);
#ifdef GEMM_ASYNC
  wait_async0();
#endif
  __syncthreads();

  int cur = 0;
  for (int k0 = 0; k0 < K; k0 += 32) {
    if (k0 + 32 < K) issue(cur ^ 1, k0 + 32);  // prefetch next tile into other buffer

    ABu af[2], bfr[2];
    const int c4 = hi * 4, c8 = hi * 8;
#pragma unroll
    for (int t = 0; t < 2; ++t) {
      const unsigned int* ar = lsA[cur][wm + t * 16 + r0];
      const unsigned int* br = lsB[cur][wn + t * 16 + r0];
#pragma unroll
      for (int p = 0; p < 8; ++p) {
        af[t].u[p]  = ar[c4 + (p < 4 ? p : 4 + p)];
        bfr[t].u[p] = br[c4 + (p < 4 ? p : 4 + p)];
      }
    }
#pragma unroll
    for (int i = 0; i < 2; ++i)
#pragma unroll
      for (int j = 0; j < 2; ++j)
        acc[i][j] = __builtin_amdgcn_wmma_f32_16x16x32_bf16(
            false, af[i].v, false, bfr[j].v, (short)0, acc[i][j], false, false);

#ifdef GEMM_ASYNC
    wait_async0();     // my async writes into next buffer have landed
#endif
    __syncthreads();   // everyone done reading cur / writing next
    cur ^= 1;
  }

  // epilogue: D frag lane l -> n = l&15, m = vgpr + (l>>4)*8
#pragma unroll
  for (int i = 0; i < 2; ++i) {
#pragma unroll
    for (int j = 0; j < 2; ++j) {
      const int n = n0 + wn + j * 16 + r0;
      const float bv = bias ? bias[n] : 0.f;
#pragma unroll
      for (int v = 0; v < 8; ++v) {
        const int m = m0 + wm + i * 16 + hi * 8 + v;
        float val = acc[i][j][v] + bv;
        if (GELU) val = 0.5f * val * (1.f + erff(val * 0.70710678118654752f));
        const size_t off = (size_t)m * N + n;
        if (Cf) Cf[off] = val;
        if (Cb) Cb[off] = (bf16_t)val;
      }
    }
  }
}

// ---------------------------------------------------------------------------
// self-attention (causal, 18 tokens): one thread per (b, h, row)
// ---------------------------------------------------------------------------
__global__ void k_self_attn(const float* __restrict__ Q, const float* __restrict__ Kx,
                            const float* __restrict__ V, bf16_t* __restrict__ out) {
  int idx = blockIdx.x * blockDim.x + threadIdx.x;
  if (idx >= Bc * Hc * NQc) return;
  const int m = idx % NQc;
  const int h = (idx / NQc) % Hc;
  const int b = idx / (NQc * Hc);

  const float* qr = Q + (size_t)(b * NQc + m) * Dc + h * DHc;
  float q[DHc];
#pragma unroll
  for (int d = 0; d < DHc; ++d) q[d] = qr[d];

  float s[NQc];
  float mx = NEGINF;
  for (int j = 0; j <= m; ++j) {
    const float* kr = Kx + (size_t)(b * NQc + j) * Dc + h * DHc;
    float a = 0.f;
#pragma unroll
    for (int d = 0; d < DHc; ++d) a += q[d] * kr[d];
    a *= SCALEc;
    s[j] = a;
    mx = fmaxf(mx, a);
  }
  float sum = 0.f;
  for (int j = 0; j <= m; ++j) { s[j] = __expf(s[j] - mx); sum += s[j]; }
  const float inv = 1.f / sum;

  float o[DHc];
#pragma unroll
  for (int d = 0; d < DHc; ++d) o[d] = 0.f;
  for (int j = 0; j <= m; ++j) {
    const float* vr = V + (size_t)(b * NQc + j) * Dc + h * DHc;
    const float a = s[j] * inv;
#pragma unroll
    for (int d = 0; d < DHc; ++d) o[d] += a * vr[d];
  }
  bf16_t* orow = out + (size_t)(b * NQc + m) * Dc + h * DHc;
#pragma unroll
  for (int d = 0; d < DHc; ++d) orow[d] = (bf16_t)o[d];
}

// ---------------------------------------------------------------------------
// cross-attention: one 256-thread block per (b,h). Full 18x1024 score matrix
// in LDS (~74KB; CDNA5 allows up to 320KB/WG). V staged in 128-key LDS chunks.
// ---------------------------------------------------------------------------
__global__ __launch_bounds__(256)
void k_cross_attn(const float* __restrict__ Q, const bf16_t* __restrict__ Kc,
                  const bf16_t* __restrict__ Vc, const unsigned char* __restrict__ kpm,
                  bf16_t* __restrict__ out) {
  __shared__ float sQ[NQc][DHc];
  __shared__ float sS[NQc][HWc];
  __shared__ unsigned int sV[128][17];   // 128 keys x 32 bf16 (u32 pairs), padded
  const int bh = blockIdx.x;
  const int b = bh >> 3, h = bh & 7;
  const int tid = threadIdx.x;

  for (int i = tid; i < NQc * DHc; i += 256) {
    int m = i >> 5, d = i & 31;
    sQ[m][d] = Q[(size_t)(b * NQc + m) * Dc + h * DHc + d];
  }
  __syncthreads();

  // scores: vectorized K row loads + bit-trick bf16->f32 unpack
  for (int j = tid; j < HWc; j += 256) {
    const uint4* kp = (const uint4*)(Kc + (size_t)(b * HWc + j) * Dc + h * DHc);
    uint4 kq[4];
    kq[0] = kp[0]; kq[1] = kp[1]; kq[2] = kp[2]; kq[3] = kp[3];
    float kv[DHc];
#pragma unroll
    for (int q4 = 0; q4 < 4; ++q4) {
      unpack2(kq[q4].x, kv + q4 * 8 + 0);
      unpack2(kq[q4].y, kv + q4 * 8 + 2);
      unpack2(kq[q4].z, kv + q4 * 8 + 4);
      unpack2(kq[q4].w, kv + q4 * 8 + 6);
    }
    const bool masked = kpm[(size_t)b * HWc + j] != 0;
#pragma unroll 1
    for (int m = 0; m < NQc; ++m) {
      float a = 0.f;
#pragma unroll
      for (int d = 0; d < DHc; ++d) a += sQ[m][d] * kv[d];
      sS[m][j] = masked ? NEGINF : a * SCALEc;
    }
  }
  __syncthreads();

  // softmax: one wave per row
  const int wave = tid >> 5, lane = tid & 31;
  for (int m = wave; m < NQc; m += 8) {
    float mx = NEGINF;
    for (int j = lane; j < HWc; j += 32) mx = fmaxf(mx, sS[m][j]);
#pragma unroll
    for (int off = 16; off; off >>= 1) mx = fmaxf(mx, __shfl_xor(mx, off, 32));
    float sum = 0.f;
    for (int j = lane; j < HWc; j += 32) {
      float e = __expf(sS[m][j] - mx);
      sS[m][j] = e;
      sum += e;
    }
#pragma unroll
    for (int off = 16; off; off >>= 1) sum += __shfl_xor(sum, off, 32);
    const float inv = 1.f / sum;
    for (int j = lane; j < HWc; j += 32) sS[m][j] *= inv;
  }
  __syncthreads();

  // out = attn @ V with V chunks staged in LDS
  float accv[3] = {0.f, 0.f, 0.f};
  int mi[3], di[3];
  int npair = 0;
  for (int i = tid; i < NQc * DHc; i += 256) { mi[npair] = i >> 5; di[npair] = i & 31; ++npair; }

  for (int c0 = 0; c0 < HWc; c0 += 128) {
    {
      const int r = tid >> 1, hf = tid & 1;
      const uint4* vp = (const uint4*)(Vc + (size_t)(b * HWc + c0 + r) * Dc + h * DHc) + hf * 2;
      const uint4 v0 = vp[0], v1 = vp[1];
      unsigned int* vs = &sV[r][hf * 8];
      vs[0]=v0.x; vs[1]=v0.y; vs[2]=v0.z; vs[3]=v0.w;
      vs[4]=v1.x; vs[5]=v1.y; vs[6]=v1.z; vs[7]=v1.w;
    }
    __syncthreads();
    for (int p = 0; p < npair; ++p) {
      const int m = mi[p], du = di[p] >> 1;
      const bool hb = di[p] & 1;
      float a = accv[p];
#pragma unroll 4
      for (int jj = 0; jj < 128; ++jj) {
        const unsigned u = sV[jj][du];
        const float vv = hb ? __uint_as_float(u & 0xffff0000u) : __uint_as_float(u << 16);
        a += sS[m][c0 + jj] * vv;
      }
      accv[p] = a;
    }
    __syncthreads();
  }
  for (int p = 0; p < npair; ++p)
    out[(size_t)(b * NQc + mi[p]) * Dc + h * DHc + di[p]] = (bf16_t)accv[p];
}

// ---------------------------------------------------------------------------
// residual + layernorm: x = LN(x + y) * w + b ; emits f32 (x) and bf16 (xb)
// one wave per 256-wide row
// ---------------------------------------------------------------------------
__global__ void k_ln(float* __restrict__ x, const float* __restrict__ y,
                     const float* __restrict__ w, const float* __restrict__ bsh,
                     bf16_t* __restrict__ xb, int rows) {
  const int row = blockIdx.x * (blockDim.x >> 5) + (threadIdx.x >> 5);
  if (row >= rows) return;
  const int lane = threadIdx.x & 31;
  float* xr = x + (size_t)row * Dc;
  const float* yr = y + (size_t)row * Dc;
  float v[8];
  float sum = 0.f;
#pragma unroll
  for (int i = 0; i < 8; ++i) { v[i] = xr[lane + 32 * i] + yr[lane + 32 * i]; sum += v[i]; }
#pragma unroll
  for (int off = 16; off; off >>= 1) sum += __shfl_xor(sum, off, 32);
  const float mu = sum * (1.f / 256.f);
  float var = 0.f;
#pragma unroll
  for (int i = 0; i < 8; ++i) { float d = v[i] - mu; var += d * d; }
#pragma unroll
  for (int off = 16; off; off >>= 1) var += __shfl_xor(var, off, 32);
  const float rs = rsqrtf(var * (1.f / 256.f) + 1e-5f);
  bf16_t* xbr = xb + (size_t)row * Dc;
#pragma unroll
  for (int i = 0; i < 8; ++i) {
    const int c = lane + 32 * i;
    const float o = (v[i] - mu) * rs * w[c] + bsh[c];
    xr[c] = o;
    xbr[c] = (bf16_t)o;
  }
}

// ---------------------------------------------------------------------------
// host
// ---------------------------------------------------------------------------
static inline void gemm(hipStream_t s, const bf16_t* A, const bf16_t* Bt,
                        const float* bias, float* Cf, bf16_t* Cb,
                        int M, int N, int K, bool gelu = false) {
  dim3 g(N / 64, M / 64);
  if (gelu) k_gemm<true ><<<g, 128, 0, s>>>(A, Bt, bias, Cf, Cb, M, N, K);
  else      k_gemm<false><<<g, 128, 0, s>>>(A, Bt, bias, Cf, Cb, M, N, K);
}

extern "C" void kernel_launch(void* const* d_in, const int* in_sizes, int n_in,
                              void* d_out, int out_size, void* d_ws, size_t ws_size,
                              hipStream_t stream) {
  (void)in_sizes; (void)n_in; (void)out_size; (void)ws_size;
  const float* query      = (const float*)d_in[0];
  const float* memory     = (const float*)d_in[1];
  const unsigned char* kpm = (const unsigned char*)d_in[2];  // bool mask
  const float* memory_pos = (const float*)d_in[3];
  const float* Wqp        = (const float*)d_in[4];
  const float* bqp        = (const float*)d_in[5];
  const float* sWq = (const float*)d_in[6];
  const float* sWk = (const float*)d_in[7];
  const float* sWv = (const float*)d_in[8];
  const float* sWp = (const float*)d_in[9];
  const float* sbp = (const float*)d_in[10];
  const float* cWq = (const float*)d_in[11];
  const float* cWk = (const float*)d_in[12];
  const float* cWv = (const float*)d_in[13];
  const float* cWp = (const float*)d_in[14];
  const float* cbp = (const float*)d_in[15];
  const float* ln1w = (const float*)d_in[16];
  const float* ln1b = (const float*)d_in[17];
  const float* ln2w = (const float*)d_in[18];
  const float* ln2b = (const float*)d_in[19];
  const float* ln3w = (const float*)d_in[20];
  const float* ln3b = (const float*)d_in[21];
  const float* fc1w = (const float*)d_in[22];
  const float* fc1b = (const float*)d_in[23];
  const float* fc2w = (const float*)d_in[24];
  const float* fc2b = (const float*)d_in[25];

  // ---- workspace carve ----
  char* p = (char*)d_ws;
  auto alloc = [&](size_t bytes) -> void* {
    void* r = (void*)p;
    p += (bytes + 255) & ~(size_t)255;
    return r;
  };
  float*  xf   = (float*) alloc((size_t)MQ * Dc * 4);
  bf16_t* xb   = (bf16_t*)alloc((size_t)MQ * Dc * 2);
  float*  qpos = (float*) alloc((size_t)MQ * Dc * 4);
  bf16_t* xq   = (bf16_t*)alloc((size_t)MQ * Dc * 2);   // also query_bf16 early
  float*  t0   = (float*) alloc((size_t)MQ * Dc * 4);
  float*  t1   = (float*) alloc((size_t)MQ * Dc * 4);
  float*  t2   = (float*) alloc((size_t)MQ * Dc * 4);
  float*  t3   = (float*) alloc((size_t)MQ * Dc * 4);
  bf16_t* attb = (bf16_t*)alloc((size_t)MQ * Dc * 2);
  bf16_t* hb   = (bf16_t*)alloc((size_t)MQ * HIDc * 2);
  bf16_t* kvb  = (bf16_t*)alloc((size_t)MK * Dc * 2);
  bf16_t* memb = (bf16_t*)alloc((size_t)MK * Dc * 2);
  bf16_t* Kb   = (bf16_t*)alloc((size_t)MK * Dc * 2);
  bf16_t* Vb   = (bf16_t*)alloc((size_t)MK * Dc * 2);
  bf16_t* WqpT = (bf16_t*)alloc((size_t)Dc * Dc * 2);
  bf16_t* sWqT = (bf16_t*)alloc((size_t)Lc * Dc * Dc * 2);
  bf16_t* sWkT = (bf16_t*)alloc((size_t)Lc * Dc * Dc * 2);
  bf16_t* sWvT = (bf16_t*)alloc((size_t)Lc * Dc * Dc * 2);
  bf16_t* sWpT = (bf16_t*)alloc((size_t)Lc * Dc * Dc * 2);
  bf16_t* cWqT = (bf16_t*)alloc((size_t)Lc * Dc * Dc * 2);
  bf16_t* cWkT = (bf16_t*)alloc((size_t)Lc * Dc * Dc * 2);
  bf16_t* cWvT = (bf16_t*)alloc((size_t)Lc * Dc * Dc * 2);
  bf16_t* cWpT = (bf16_t*)alloc((size_t)Lc * Dc * Dc * 2);
  bf16_t* fc1T = (bf16_t*)alloc((size_t)Lc * HIDc * Dc * 2);  // [l][HID][D]
  bf16_t* fc2T = (bf16_t*)alloc((size_t)Lc * Dc * HIDc * 2);  // [l][D][HID]

  const int nMQD = MQ * Dc;
  const int nMKD = MK * Dc;
  const dim3 tb(32, 8);

  // ---- one-time conversions ----
  k_cvt4<<<(nMQD / 4 + 255) / 256, 256, 0, stream>>>(xq, query, nMQD / 4);
  k_transpose<<<dim3(Dc / 32, Dc / 32, 1),  tb, 0, stream>>>(WqpT, Wqp, Dc, Dc);
  k_transpose<<<dim3(Dc / 32, Dc / 32, Lc), tb, 0, stream>>>(sWqT, sWq, Dc, Dc);
  k_transpose<<<dim3(Dc / 32, Dc / 32, Lc), tb, 0, stream>>>(sWkT, sWk, Dc, Dc);
  k_transpose<<<dim3(Dc / 32, Dc / 32, Lc), tb, 0, stream>>>(sWvT, sWv, Dc, Dc);
  k_transpose<<<dim3(Dc / 32, Dc / 32, Lc), tb, 0, stream>>>(sWpT, sWp, Dc, Dc);
  k_transpose<<<dim3(Dc / 32, Dc / 32, Lc), tb, 0, stream>>>(cWqT, cWq, Dc, Dc);
  k_transpose<<<dim3(Dc / 32, Dc / 32, Lc), tb, 0, stream>>>(cWkT, cWk, Dc, Dc);
  k_transpose<<<dim3(Dc / 32, Dc / 32, Lc), tb, 0, stream>>>(cWvT, cWv, Dc, Dc);
  k_transpose<<<dim3(Dc / 32, Dc / 32, Lc), tb, 0, stream>>>(cWpT, cWp, Dc, Dc);
  k_transpose<<<dim3(HIDc / 32, Dc / 32, Lc), tb, 0, stream>>>(fc1T, fc1w, Dc, HIDc);
  k_transpose<<<dim3(Dc / 32, HIDc / 32, Lc), tb, 0, stream>>>(fc2T, fc2w, HIDc, Dc);

  k_add_cvt4<<<(nMKD / 4 + 255) / 256, 256, 0, stream>>>(kvb, memory, memory_pos, nMKD / 4);
  k_cvt4<<<(nMKD / 4 + 255) / 256, 256, 0, stream>>>(memb, memory, nMKD / 4);

  // qpos = query @ Wqp + bqp
  gemm(stream, xq, WqpT, bqp, qpos, nullptr, MQ, Dc, Dc);

  // x = query
  (void)hipMemcpyAsync(xf, query, (size_t)nMQD * 4, hipMemcpyDeviceToDevice, stream);
  k_cvt4<<<(nMQD / 4 + 255) / 256, 256, 0, stream>>>(xb, query, nMQD / 4);

  // ---- layers ----
  for (int l = 0; l < Lc; ++l) {
    const size_t wo = (size_t)l * Dc * Dc;
    const size_t ho = (size_t)l * Dc * HIDc;
    // self-attention
    k_add_cvt4<<<(nMQD / 4 + 255) / 256, 256, 0, stream>>>(xq, xf, qpos, nMQD / 4);
    gemm(stream, xq, sWqT + wo, nullptr, t0, nullptr, MQ, Dc, Dc);
    gemm(stream, xq, sWkT + wo, nullptr, t1, nullptr, MQ, Dc, Dc);
    gemm(stream, xb, sWvT + wo, nullptr, t2, nullptr, MQ, Dc, Dc);
    k_self_attn<<<(Bc * Hc * NQc + 255) / 256, 256, 0, stream>>>(t0, t1, t2, attb);
    gemm(stream, attb, sWpT + wo, sbp + l * Dc, t3, nullptr, MQ, Dc, Dc);
    k_ln<<<MQ / 8, 256, 0, stream>>>(xf, t3, ln1w + l * Dc, ln1b + l * Dc, xb, MQ);

    // cross-attention
    k_add_cvt4<<<(nMQD / 4 + 255) / 256, 256, 0, stream>>>(xq, xf, qpos, nMQD / 4);
    gemm(stream, xq,   cWqT + wo, nullptr, t0, nullptr, MQ, Dc, Dc);
    gemm(stream, kvb,  cWkT + wo, nullptr, nullptr, Kb, MK, Dc, Dc);
    gemm(stream, memb, cWvT + wo, nullptr, nullptr, Vb, MK, Dc, Dc);
    k_cross_attn<<<Bc * Hc, 256, 0, stream>>>(t0, Kb, Vb, kpm, attb);
    gemm(stream, attb, cWpT + wo, cbp + l * Dc, t3, nullptr, MQ, Dc, Dc);
    k_ln<<<MQ / 8, 256, 0, stream>>>(xf, t3, ln2w + l * Dc, ln2b + l * Dc, xb, MQ);

    // MLP
    gemm(stream, xb, fc1T + ho, fc1b + l * HIDc, nullptr, hb, MQ, HIDc, Dc, /*gelu=*/true);
    gemm(stream, hb, fc2T + ho, fc2b + l * Dc, t3, nullptr, MQ, Dc, HIDc);
    k_ln<<<MQ / 8, 256, 0, stream>>>(xf, t3, ln3w + l * Dc, ln3b + l * Dc, xb, MQ);
  }

  (void)hipMemcpyAsync(d_out, xf, (size_t)nMQD * 4, hipMemcpyDeviceToDevice, stream);
}
